// self_LSTM_sparse_attn_predict_82497731822244
// MI455X (gfx1250) — matrix operand order repaired and checked by reading, not compiled, not run on hardware
//
#include <hip/hip_runtime.h>
#include <hip/hip_bf16.h>

// Persistent LSTM + sparse-top5-attention scan for MI455X (gfx1250).
// 2 workgroups x 256 threads (8 wave32). Each workgroup owns 16 batches.
// Weights converted to f16 once into LDS (WMMA B-fragment layout), gate GEMM
// (16x192)@(192x512) per timestep via v_wmma_f32_16x16x32_f16.
// x_t is register-prefetched one step ahead; B fragments double-buffered.
// d_ws must hold h_old: 32*257*128 f32 = ~4.2 MB.

typedef __attribute__((ext_vector_type(16))) _Float16 v16h;
typedef __attribute__((ext_vector_type(8)))  float    v8f;

#define B_TOT  32
#define T_SEQ  256
#define IN_DIM 64
#define H_DIM  128
#define G_DIM  512
#define BB     16          // batches per workgroup
#define TOPK   5
#define EPS_F  1e-7f

#define SBS    264         // score-cache row stride (floats)

// ---- dynamic LDS layout (bytes), all 16B aligned where vector-accessed ----
#define OFF_WFRAG  0                       // 32nt*6kt*32lane*16half f16 = 196608
#define OFF_XH     196608                  // 16*64 f16   = 2048
#define OFF_H16    198656                  // 16*128 f16  = 4096
#define OFF_G      202752                  // 16*512 f32  = 32768
#define OFF_SB     235520                  // 16*SBS f32  = 16896
#define OFF_SA     252416                  // 16 f32
#define OFF_BIAS   252480                  // 512 f32
#define OFF_WA     254528                  // 128 f32
#define OFF_WB     255040                  // 128 f32
#define OFF_T5V    255552                  // 16*16*5 f32 = 5120
#define OFF_T5I    260672                  // 16*16*5 i32 = 5120
#define OFF_ESEL   265792                  // 16*5 f32
#define OFF_EIDX   266112                  // 16*5 i32
#define OFF_RED    266432                  // 16*16 f32
#define OFF_DELTA  267456                  // 16 f32
#define OFF_INVN   267520                  // 16 f32
#define SMEM_BYTES 267584

union F16x16 { v16h v; uint4 q[2]; };
union H4Pack { _Float16 h[4]; uint2 u; };

__device__ __forceinline__ float sigm(float x) { return 1.0f / (1.0f + __expf(-x)); }

__device__ __forceinline__ void insert5(float v[5], int ix[5], float nv, int ni) {
#pragma unroll
  for (int p = 0; p < 5; ++p) {
    if (nv > v[p]) {
#pragma unroll
      for (int q = 4; q > p; --q) { v[q] = v[q-1]; ix[q] = ix[q-1]; }
      v[p] = nv; ix[p] = ni;
      break;
    }
  }
}

extern __shared__ char smem[];

__global__ __launch_bounds__(256, 1)
void lstm_sparse_attn_kernel(const float* __restrict__ x,
                             const float* __restrict__ W_ih,
                             const float* __restrict__ W_hh,
                             const float* __restrict__ b_ih,
                             const float* __restrict__ b_hh,
                             const float* __restrict__ w_t,
                             float* __restrict__ out,
                             float* __restrict__ h_old) {
  const int tid  = threadIdx.x;
  const int lane = tid & 31;
  const int wave = tid >> 5;
  const int boff = blockIdx.x * BB;

  _Float16* WFRAG = (_Float16*)(smem + OFF_WFRAG);
  _Float16* XH    = (_Float16*)(smem + OFF_XH);
  _Float16* H16   = (_Float16*)(smem + OFF_H16);
  float*    G     = (float*)(smem + OFF_G);
  float*    SB    = (float*)(smem + OFF_SB);
  float*    SA    = (float*)(smem + OFF_SA);
  float*    BIAS  = (float*)(smem + OFF_BIAS);
  float*    WA    = (float*)(smem + OFF_WA);
  float*    WB    = (float*)(smem + OFF_WB);
  float*    T5V   = (float*)(smem + OFF_T5V);
  int*      T5I   = (int*)(smem + OFF_T5I);
  float*    ESEL  = (float*)(smem + OFF_ESEL);
  int*      EIDX  = (int*)(smem + OFF_EIDX);
  float*    RED   = (float*)(smem + OFF_RED);
  float*    DELTA = (float*)(smem + OFF_DELTA);
  float*    INVN  = (float*)(smem + OFF_INVN);

  // ---------------- init: pack weights into WMMA B-fragment layout ----------
  // B tile (K=32 x N=16) f16: lane L -> N = L&15, K = (L>>4)*16 + h, h=0..15.
  for (int e = tid; e < 32 * 6 * 32; e += 256) {
    int nt   = e / (6 * 32);
    int r    = e % (6 * 32);
    int kt   = r >> 5;
    int ln   = r & 31;
    int N    = nt * 16 + (ln & 15);
    int Kb   = kt * 32 + ((ln >> 4) << 4);
#pragma unroll
    for (int h = 0; h < 16; ++h) {
      int K = Kb + h;
      float v = (K < IN_DIM) ? W_ih[N * IN_DIM + K] : W_hh[N * H_DIM + (K - IN_DIM)];
      WFRAG[e * 16 + h] = (_Float16)v;
    }
  }
  for (int n = tid; n < G_DIM; n += 256) BIAS[n] = b_ih[n] + b_hh[n];
  for (int j = tid; j < H_DIM; j += 256) { WA[j] = w_t[j]; WB[j] = w_t[H_DIM + j]; }
  for (int e = tid; e < BB * H_DIM; e += 256) H16[e] = (_Float16)0.0f;
  for (int e = tid; e < BB * SBS; e += 256) SB[e] = 0.0f;
  for (int e = tid; e < BB * H_DIM; e += 256) {           // h_old row 0 = zeros
    int bb = e >> 7, j = e & 127;
    h_old[(((size_t)(boff + bb)) * (T_SEQ + 1) + 0) * H_DIM + j] = 0.0f;
  }

  float creg[8];                                          // c state, per-thread
#pragma unroll
  for (int k = 0; k < 8; ++k) creg[k] = 0.0f;
  const int b   = tid >> 4;
  const int sub = tid & 15;
  const int j0  = sub * 8;

  // x prefetch: thread owns 4 contiguous inputs of batch b: k4 = sub*4
  const float* xbase = x + ((size_t)(boff + b)) * T_SEQ * IN_DIM + sub * 4;
  float4 xreg = *(const float4*)(xbase);                  // step 0

  const v8f vzero = {};

  __syncthreads();

  // ---------------- sequential scan ----------------------------------------
  for (int step = 0; step < T_SEQ; ++step) {
    const int rem = step + 1;

    // commit prefetched x_t as f16 (single ds_store_b64 per thread)
    {
      H4Pack pk;
      pk.h[0] = (_Float16)xreg.x; pk.h[1] = (_Float16)xreg.y;
      pk.h[2] = (_Float16)xreg.z; pk.h[3] = (_Float16)xreg.w;
      *(uint2*)(XH + b * IN_DIM + sub * 4) = pk.u;
    }
    __syncthreads();

    // issue next step's x load now; consumed next iteration (latency hidden)
    if (step + 1 < T_SEQ) xreg = *(const float4*)(xbase + (size_t)(step + 1) * IN_DIM);

    // A fragments: 16x32 f16, lane L -> M = L&15, K = (L>>4)*16 + h
    F16x16 afrag[6];
    {
      const int M  = lane & 15;
      const int hi = (lane >> 4) * 16;
#pragma unroll
      for (int kt = 0; kt < 2; ++kt) {
        const _Float16* p = XH + M * IN_DIM + kt * 32 + hi;
        afrag[kt].q[0] = *(const uint4*)p;
        afrag[kt].q[1] = *(const uint4*)(p + 8);
      }
#pragma unroll
      for (int kt = 2; kt < 6; ++kt) {
        const _Float16* p = H16 + M * H_DIM + (kt - 2) * 32 + hi;
        afrag[kt].q[0] = *(const uint4*)p;
        afrag[kt].q[1] = *(const uint4*)(p + 8);
      }
    }

    // GEMM: wave owns 4 N-tiles; flat 24-iter loop, double-buffered B frags.
    // B frag address stride per iteration is exactly 512 halfs (1024 B).
    {
      const _Float16* wbase = WFRAG + ((size_t)wave * 24 * 32 + lane) * 16;
      F16x16 bf[2];
      bf[0].q[0] = *(const uint4*)(wbase);
      bf[0].q[1] = *(const uint4*)(wbase + 8);
      v8f acc = vzero;
#pragma unroll
      for (int it = 0; it < 24; ++it) {
        const int cur = it & 1;
        if (it + 1 < 24) {
          const _Float16* bp = wbase + (size_t)(it + 1) * 512;
          bf[cur ^ 1].q[0] = *(const uint4*)bp;
          bf[cur ^ 1].q[1] = *(const uint4*)(bp + 8);
        }
        const int kt = it % 6;
        acc = __builtin_amdgcn_wmma_f32_16x16x32_f16(
            false, afrag[kt].v, false, bf[cur].v, (short)0, acc, false, false);
        if (kt == 5) {
          const int nt = wave * 4 + it / 6;
          const int Mb = (lane >> 4) * 8;
          const int n  = nt * 16 + (lane & 15);
#pragma unroll
          for (int r = 0; r < 8; ++r) G[(Mb + r) * G_DIM + n] = acc[r];
          acc = vzero;
        }
      }
    }
    __syncthreads();

    // elementwise LSTM cell + s_a partial
    float hc[8];
    float sa_part = 0.0f;
#pragma unroll
    for (int k = 0; k < 8; ++k) {
      const int j = j0 + k;
      float gi = G[b * G_DIM + j]           + BIAS[j];
      float gf = G[b * G_DIM + 128 + j]     + BIAS[128 + j];
      float gg = G[b * G_DIM + 256 + j]     + BIAS[256 + j];
      float go = G[b * G_DIM + 384 + j]     + BIAS[384 + j];
      float c  = sigm(gf) * creg[k] + sigm(gi) * tanhf(gg);
      creg[k]  = c;
      float h  = sigm(go) * tanhf(c);
      hc[k]    = h;
      sa_part += tanhf(h) * WA[j];
    }
    RED[b * 16 + sub] = sa_part;
    __syncthreads();
    if (sub == 0) {
      float s = 0.0f;
      for (int k = 0; k < 16; ++k) s += RED[b * 16 + k];
      SA[b] = s;
    }
    __syncthreads();

    // top-5 over scores s[t] = s_a + s_b[t], t < rem
    const float sab = SA[b];
    float tv[5]; int ti[5];
#pragma unroll
    for (int k = 0; k < 5; ++k) { tv[k] = -3e38f; ti[k] = 0; }
    for (int t = sub; t < rem; t += 16) insert5(tv, ti, sab + SB[b * SBS + t], t);
#pragma unroll
    for (int k = 0; k < 5; ++k) {
      T5V[(b * 16 + sub) * 5 + k] = tv[k];
      T5I[(b * 16 + sub) * 5 + k] = ti[k];
    }
    __syncthreads();
    // stage 1: 4 threads each merge 4 partial lists
    if (sub < 4) {
      float mv[5]; int mi[5];
#pragma unroll
      for (int k = 0; k < 5; ++k) { mv[k] = -3e38f; mi[k] = 0; }
      for (int g2 = 0; g2 < 4; ++g2)
        for (int k = 0; k < 5; ++k)
          insert5(mv, mi, T5V[(b * 16 + sub * 4 + g2) * 5 + k],
                          T5I[(b * 16 + sub * 4 + g2) * 5 + k]);
#pragma unroll
      for (int k = 0; k < 5; ++k) {
        T5V[(b * 16 + sub) * 5 + k] = mv[k];
        T5I[(b * 16 + sub) * 5 + k] = mi[k];
      }
    }
    __syncthreads();
    // stage 2: final merge + sparse weight computation
    if (sub == 0) {
      float mv[5]; int mi[5];
#pragma unroll
      for (int k = 0; k < 5; ++k) { mv[k] = -3e38f; mi[k] = 0; }
      for (int g2 = 0; g2 < 4; ++g2)
        for (int k = 0; k < 5; ++k)
          insert5(mv, mi, T5V[(b * 16 + g2) * 5 + k], T5I[(b * 16 + g2) * 5 + k]);
      if (rem <= TOPK) {
        for (int k = 0; k < 5; ++k) {
          ESEL[b * 5 + k] = (k < rem) ? (sab + SB[b * SBS + k]) : 0.0f;
          EIDX[b * 5 + k] = (k < rem) ? k : 0;
        }
        DELTA[b] = 0.0f; INVN[b] = 0.0f;
      } else {
        float delta = mv[4] + EPS_F;
        float nrm = 0.0f, w[5];
#pragma unroll
        for (int k = 0; k < 5; ++k) { w[k] = fmaxf(mv[k] - delta, 0.0f); nrm += w[k]; }
        float inv = 1.0f / (nrm + EPS_F);
#pragma unroll
        for (int k = 0; k < 5; ++k) { ESEL[b * 5 + k] = w[k] * inv; EIDX[b * 5 + k] = mi[k]; }
        DELTA[b] = delta; INVN[b] = inv;
      }
    }
    __syncthreads();

    // sparse attention gather (<=5 rows) + state update
    float acc2[8];
#pragma unroll
    for (int k = 0; k < 8; ++k) acc2[k] = 0.0f;
#pragma unroll
    for (int k = 0; k < 5; ++k) {
      const float w = ESEL[b * 5 + k];
      const int   t = EIDX[b * 5 + k];
      const float* row = h_old + (((size_t)(boff + b)) * (T_SEQ + 1) + t) * H_DIM + j0;
      float4 r0 = *(const float4*)row;
      float4 r1 = *(const float4*)(row + 4);
      acc2[0] += w * r0.x; acc2[1] += w * r0.y; acc2[2] += w * r0.z; acc2[3] += w * r0.w;
      acc2[4] += w * r1.x; acc2[5] += w * r1.y; acc2[6] += w * r1.z; acc2[7] += w * r1.w;
    }
    float sb_part = 0.0f;
    float* nrow = h_old + (((size_t)(boff + b)) * (T_SEQ + 1) + rem) * H_DIM + j0;
#pragma unroll
    for (int k = 0; k < 8; ++k) {
      const int j = j0 + k;
      float hn = hc[k] + acc2[k];
      H16[b * H_DIM + j] = (_Float16)hn;
      nrow[k] = hn;
      sb_part += tanhf(hn) * WB[j];
    }
    RED[b * 16 + sub] = sb_part;

    if (step == T_SEQ - 1) {
      // out[0:4096] = attn_c (B,H); out[4096:] = attn_w (B,T)
#pragma unroll
      for (int k = 0; k < 8; ++k) out[(boff + b) * H_DIM + j0 + k] = acc2[k];
      const float delta = DELTA[b], inv = INVN[b];
      for (int t = sub; t < T_SEQ; t += 16) {
        float v = sab + SB[b * SBS + t];
        out[B_TOT * H_DIM + (boff + b) * T_SEQ + t] = fmaxf(v - delta, 0.0f) * inv;
      }
    }
    __syncthreads();
    if (sub == 0) {
      float s = 0.0f;
      for (int k = 0; k < 16; ++k) s += RED[b * 16 + k];
      SB[b * SBS + rem] = s;
    }
    __syncthreads();
  }
}

extern "C" void kernel_launch(void* const* d_in, const int* in_sizes, int n_in,
                              void* d_out, int out_size, void* d_ws, size_t ws_size,
                              hipStream_t stream) {
  (void)in_sizes; (void)n_in; (void)out_size; (void)ws_size;
  const float* x    = (const float*)d_in[0];
  const float* W_ih = (const float*)d_in[1];
  const float* W_hh = (const float*)d_in[2];
  const float* b_ih = (const float*)d_in[3];
  const float* b_hh = (const float*)d_in[4];
  const float* w_t  = (const float*)d_in[5];
  float* out   = (float*)d_out;
  float* h_old = (float*)d_ws;   // 32*257*128 f32 ~ 4.2 MB

  (void)hipFuncSetAttribute((const void*)lstm_sparse_attn_kernel,
                            hipFuncAttributeMaxDynamicSharedMemorySize,
                            SMEM_BYTES);
  lstm_sparse_attn_kernel<<<2, 256, SMEM_BYTES, stream>>>(
      x, W_ih, W_hh, b_ih, b_hh, w_t, out, h_old);
}
